// EnhancedSNN_47313359732883
// MI455X (gfx1250) — matrix operand order repaired and checked by reading, not compiled, not run on hardware
//
#include <hip/hip_runtime.h>
#include <hip/hip_bf16.h>

typedef _Float16 v16h __attribute__((ext_vector_type(16)));
typedef _Float16 v8h  __attribute__((ext_vector_type(8)));
typedef _Float16 v4h  __attribute__((ext_vector_type(4)));
typedef float    v8f  __attribute__((ext_vector_type(8)));

#define NB     2048
#define NI     1024
#define NH     1024
#define NO     10
#define TSTEPS 50
#define BETA   0.9f

#define SPK1_TOTAL ((size_t)TSTEPS * NB * NH)

// ---------------------------------------------------------------------------
// f32 -> f16 conversion (4 elements/thread, vectorized load/store)
// ---------------------------------------------------------------------------
__global__ __launch_bounds__(256) void cvt_f32_to_f16_x4(const float* __restrict__ in,
                                                         _Float16* __restrict__ out,
                                                         int n4) {
  int i = blockIdx.x * 256 + threadIdx.x;
  if (i < n4) {
    float4 v = ((const float4*)in)[i];
    v4h h = { (_Float16)v.x, (_Float16)v.y, (_Float16)v.z, (_Float16)v.w };
    *(v4h*)(out + (size_t)i * 4) = h;
  }
}

// ---------------------------------------------------------------------------
// Fragment loaders (contiguous k order used by this code, same for A and B)
// A 16x32 f16: lanes 0-15 (M=lane): K = k0+0..15 (contiguous)
//              lanes 16-31 (M=lane-16): K = k0+16..31
// B 32x16 f16: N = lane&15; lanes 0-15 hold K=k0..k0+15 (contiguous),
//              lanes 16-31 hold K=k0+16..k0+31. w1 row n == B column n.
// ---------------------------------------------------------------------------
__device__ __forceinline__ v16h load_a_frag(const _Float16* __restrict__ arow,
                                            int k0, int hb) {
  v8h lo = *(const v8h*)(arow + k0 + hb * 16);
  v8h hi = *(const v8h*)(arow + k0 + hb * 16 + 8);
  v16h a;
#pragma unroll
  for (int i = 0; i < 8; ++i) { a[i] = lo[i]; a[8 + i] = hi[i]; }
  return a;
}

__device__ __forceinline__ v16h load_b_frag(const _Float16* __restrict__ bcolbase,
                                            int k0) {
  return *(const v16h*)(bcolbase + k0);   // 32B contiguous per lane
}

// ---------------------------------------------------------------------------
// fc1: cur1[m][n] = sum_k x[m][k] * w1[n][k] + b1[n]
// One wave computes a 16x64 tile (4 accumulators, A reused).
// Ping-pong software pipeline: two fragment register sets alternate across
// half-iterations, so loop-carried values are defined directly by loads and
// consumed by WMMAs — no rotation copies.
// ---------------------------------------------------------------------------
__global__ __launch_bounds__(256) void fc1_wmma(const _Float16* __restrict__ xh,
                                                const _Float16* __restrict__ w1h,
                                                const float* __restrict__ b1,
                                                float* __restrict__ cur1) {
  const int lane = threadIdx.x & 31;
  const int wid  = threadIdx.x >> 5;
  const int gw   = blockIdx.x * 8 + wid;   // 0 .. 2047
  const int mtile = gw >> 4;               // NB/16 = 128 row tiles
  const int ntile = gw & 15;               // NH/64 = 16 col tiles
  const int m0 = mtile * 16;
  const int n0 = ntile * 64;
  const int hb = lane >> 4;
  const int l  = lane & 15;

  const _Float16* __restrict__ arow = xh + (size_t)(m0 + l) * NI;
  const _Float16* bcol[4];
#pragma unroll
  for (int s = 0; s < 4; ++s)
    bcol[s] = w1h + (size_t)(n0 + s * 16 + l) * NI + hb * 16;

  v8f acc[4] = {};

  v16h aA, aB;
  v16h bA[4], bB[4];

  auto load_frags = [&](int k0, v16h& a, v16h* b) {
    a = load_a_frag(arow, k0, hb);
#pragma unroll
    for (int s = 0; s < 4; ++s) b[s] = load_b_frag(bcol[s], k0);
  };
  auto compute = [&](const v16h& a, const v16h* b) {
#pragma unroll
    for (int s = 0; s < 4; ++s)
      acc[s] = __builtin_amdgcn_wmma_f32_16x16x32_f16(
          false, a, false, b[s], (short)0, acc[s], false, false);
  };

  // prologue
  load_frags(0, aA, bA);

  int k0 = 0;
  for (; k0 + 64 < NI; k0 += 64) {
    load_frags(k0 + 32, aB, bB);   // prefetch odd chunk
    compute(aA, bA);               // consume even chunk
    load_frags(k0 + 64, aA, bA);   // prefetch next even chunk
    compute(aB, bB);               // consume odd chunk
  }
  // epilogue: k0 == NI - 64
  load_frags(NI - 32, aB, bB);
  compute(aA, bA);
  compute(aB, bB);

  // ---- store with bias (C/D: VGPR r -> M = r + hb*8, N = l)
#pragma unroll
  for (int s = 0; s < 4; ++s) {
    const int n = n0 + s * 16 + l;
    const float bias = b1[n];
#pragma unroll
    for (int r = 0; r < 8; ++r) {
      const int m = m0 + hb * 8 + r;
      cur1[(size_t)m * NH + n] = acc[s][r] + bias;
    }
  }
}

// ---------------------------------------------------------------------------
// Fused 50-step LIF recurrence.
// One block per batch row. mem1 lives in registers (4 cols/thread),
// w2 staged in LDS once, cur2 via wave32 shuffle + cross-wave LDS reduction,
// mem2 in registers of lanes 0..9. Only global traffic per step = spike writes.
// ---------------------------------------------------------------------------
__global__ __launch_bounds__(256) void snn_steps(const float* __restrict__ cur1,
                                                 const float* __restrict__ w2,
                                                 const float* __restrict__ b2,
                                                 float* __restrict__ out) {
  __shared__ float w2s[NO * NH];   // 40 KB (of 320 KB/WGP)
  __shared__ float red[8 * NO];    // per-wave partial cur2

  const int row = blockIdx.x;
  const int tid = threadIdx.x;

  for (int i = tid; i < NO * NH; i += 256) w2s[i] = w2[i];

  float c1[4], m1[4];
#pragma unroll
  for (int j = 0; j < 4; ++j) {
    c1[j] = cur1[(size_t)row * NH + tid + j * 256];
    m1[j] = 0.0f;
  }
  float mem2 = 0.0f;
  const float b2v = (tid < NO) ? b2[tid] : 0.0f;
  __syncthreads();

  for (int t = 0; t < TSTEPS; ++t) {
    float part[NO];
#pragma unroll
    for (int o = 0; o < NO; ++o) part[o] = 0.0f;

    float* __restrict__ out1 = out + (size_t)t * NB * NH + (size_t)row * NH;
#pragma unroll
    for (int j = 0; j < 4; ++j) {
      const int h = tid + j * 256;
      const float r1 = (m1[j] > 1.0f) ? 1.0f : 0.0f;  // reset from prev mem1
      m1[j] = BETA * m1[j] + c1[j] - r1;
      const float s1 = (m1[j] > 1.0f) ? 1.0f : 0.0f;  // spike
      out1[h] = s1;                                    // coalesced
#pragma unroll
      for (int o = 0; o < NO; ++o)
        part[o] = fmaf(s1, w2s[o * NH + h], part[o]);
    }

    // wave32 reduction of the 10 partial dot products
#pragma unroll
    for (int off = 16; off > 0; off >>= 1) {
#pragma unroll
      for (int o = 0; o < NO; ++o)
        part[o] += __shfl_down(part[o], off, 32);
    }
    if ((tid & 31) == 0) {
#pragma unroll
      for (int o = 0; o < NO; ++o) red[(tid >> 5) * NO + o] = part[o];
    }
    __syncthreads();

    if (tid < NO) {
      float c2 = b2v;
#pragma unroll
      for (int w = 0; w < 8; ++w) c2 += red[w * NO + tid];
      const float r2 = (mem2 > 1.0f) ? 1.0f : 0.0f;
      mem2 = BETA * mem2 + c2 - r2;
      const float s2 = (mem2 > 1.0f) ? 1.0f : 0.0f;
      out[SPK1_TOTAL + (size_t)t * NB * NO + (size_t)row * NO + tid] = s2;
    }
    __syncthreads();  // guard red[] reuse next step
  }
}

// ---------------------------------------------------------------------------
extern "C" void kernel_launch(void* const* d_in, const int* in_sizes, int n_in,
                              void* d_out, int out_size, void* d_ws, size_t ws_size,
                              hipStream_t stream) {
  const float* x  = (const float*)d_in[0];  // [2048,1024]
  const float* w1 = (const float*)d_in[1];  // [1024,1024]
  const float* b1 = (const float*)d_in[2];  // [1024]
  const float* w2 = (const float*)d_in[3];  // [10,1024]
  const float* b2 = (const float*)d_in[4];  // [10]
  float* out = (float*)d_out;

  // workspace layout: xh (4MB) | w1h (2MB) | cur1 (8MB)
  _Float16* xh  = (_Float16*)d_ws;
  _Float16* w1h = xh + (size_t)NB * NI;
  float*    cur1 = (float*)(w1h + (size_t)NH * NI);

  {
    int n4 = (NB * NI) / 4;
    cvt_f32_to_f16_x4<<<(n4 + 255) / 256, 256, 0, stream>>>(x, xh, n4);
  }
  {
    int n4 = (NH * NI) / 4;
    cvt_f32_to_f16_x4<<<(n4 + 255) / 256, 256, 0, stream>>>(w1, w1h, n4);
  }

  // (NB/16)*(NH/64) = 2048 waves -> 256 blocks of 8 waves
  fc1_wmma<<<256, 256, 0, stream>>>(xh, w1h, b1, cur1);

  snn_steps<<<NB, 256, 0, stream>>>(cur1, w2, b2, out);
}
